// RGCN_10977936409132
// MI455X (gfx1250) — matrix operand order repaired and checked by reading, not compiled, not run on hardware
//
#include <hip/hip_runtime.h>

// ---------------------------------------------------------------------------
// RGCN-LSTM on MI455X (gfx1250): bf16 WMMA (v_wmma_f32_16x16x32_bf16),
// f32 accumulate.  A (64MB f32 -> 32MB bf16) stays L2-resident; per-step
// working set ~60MB < 192MB L2, so the big A@XG GEMM (17.2 GFLOP/step) is
// matrix-core bound, not HBM bound.  GEMM mainloops use a double-buffered
// LDS pipeline fed by GLOBAL_LOAD_ASYNC_TO_LDS_B128 (ASYNCcnt) when the
// toolchain exposes the builtins, so tile fetch overlaps the WMMAs.
// ---------------------------------------------------------------------------

#define NN 4096
#define TT 365
#define DD 16
#define HH 256

typedef unsigned short u16;
typedef unsigned int   u32;
typedef __attribute__((ext_vector_type(16))) __bf16 v16bf;
typedef __attribute__((ext_vector_type(8)))  float  v8f;

// b128 payload type exactly matching the async-builtin signature:
// "__attribute__((__vector_size__(4 * sizeof(int)))) int"
typedef int v4i_vs __attribute__((vector_size(16)));
typedef __attribute__((address_space(1))) v4i_vs* gas_v4i;   // global
typedef __attribute__((address_space(3))) v4i_vs* las_v4i;   // LDS

#define LDS_STRIDE 40            // 32 bf16 + 8 pad; 80B rows keep 16B chunks aligned
#define TILE_E (128 * LDS_STRIDE)

// ---- CDNA5 async copy to LDS (ASYNCcnt path), guarded by __has_builtin ----
#if defined(__has_builtin)
#if __has_builtin(__builtin_amdgcn_global_load_async_to_lds_b128) && \
    __has_builtin(__builtin_amdgcn_s_wait_asynccnt)
#define USE_ASYNC_LDS 1
#endif
#endif
#ifndef USE_ASYNC_LDS
#define USE_ASYNC_LDS 0
#endif

__device__ __forceinline__ void async_wait_all() {
#if USE_ASYNC_LDS
  __builtin_amdgcn_s_wait_asynccnt(0);
#endif
}

__device__ __forceinline__ u16 f2bf(float f) {   // round-to-nearest-even
  u32 u = __float_as_uint(f);
  u += 0x7FFFu + ((u >> 16) & 1u);
  return (u16)(u >> 16);
}
__device__ __forceinline__ float bf2f(u16 s) {
  return __uint_as_float(((u32)s) << 16);
}
__device__ __forceinline__ float sigmoidf_(float x) {
  return 1.0f / (1.0f + __expf(-x));
}

union FragU { uint4 u[2]; v16bf v; };

// A fragment 16x32 bf16 (ISA 7.12.2): row M = lane&15; lanes<16 hold
// K[0:8)+K[16:24), lanes>=16 hold K[8:16)+K[24:32).
__device__ __forceinline__ v16bf frag_a(const u16* lds, int row, int lane) {
  const int half = (lane >> 4) * 8;
  FragU r;
  r.u[0] = *(const uint4*)(lds + row * LDS_STRIDE + half);
  r.u[1] = *(const uint4*)(lds + row * LDS_STRIDE + half + 16);
  return r.v;
}
// B fragment 32x16 bf16, tile stored N-major in LDS: col N = lane&15;
// lanes<16 hold K[0:8)+K[16:24), lanes>=16 hold K[8:16)+K[24:32).
__device__ __forceinline__ v16bf frag_b(const u16* lds, int col, int lane) {
  const int half = (lane >> 4) * 8;
  FragU r;
  r.u[0] = *(const uint4*)(lds + col * LDS_STRIDE + half);
  r.u[1] = *(const uint4*)(lds + col * LDS_STRIDE + half + 16);
  return r.v;
}

// Stage a 128x32 bf16 tile (row-major, leading dim ld elems) into LDS.
// 256 threads x 2 x 16B chunks.  Async variant: global_load_async_to_lds_b128.
__device__ __forceinline__ void fetch_tile(u16* lds, const u16* g, int ld,
                                           int row0, int k0, int tid) {
#pragma unroll
  for (int j = 0; j < 2; ++j) {
    const int c = tid + j * 256;           // 512 chunks of 8 elems
    const int row = c >> 2, seg = (c & 3) * 8;
    const u16* src = g + (size_t)(row0 + row) * ld + k0 + seg;
    u16* dst = lds + row * LDS_STRIDE + seg;
#if USE_ASYNC_LDS
    __builtin_amdgcn_global_load_async_to_lds_b128(
        (gas_v4i)(unsigned long long)(const void*)src,
        (las_v4i)(unsigned)(unsigned long long)(void*)dst,
        0, 0);
#else
    *(uint4*)dst = *(const uint4*)src;
#endif
  }
}

// One K=32 chunk of the 128x128 block tile: wave grid 2(M) x 4(N),
// each wave 64x32 = 4x2 WMMA tiles.
__device__ __forceinline__ void mma_tile(const u16* sA, const u16* sB,
                                         v8f (&acc)[4][2], int lane,
                                         int waveM, int waveN) {
  v16bf b[2];
#pragma unroll
  for (int ni = 0; ni < 2; ++ni)
    b[ni] = frag_b(sB, waveN * 32 + ni * 16 + (lane & 15), lane);
#pragma unroll
  for (int mi = 0; mi < 4; ++mi) {
    v16bf a = frag_a(sA, waveM * 64 + mi * 16 + (lane & 15), lane);
#pragma unroll
    for (int ni = 0; ni < 2; ++ni)
      acc[mi][ni] = __builtin_amdgcn_wmma_f32_16x16x32_bf16(
          false, a, false, b[ni], (short)0, acc[mi][ni], false, false);
  }
}

// Double-buffered mainloop: one barrier per K-chunk; next tile's copies are
// issued before this chunk's WMMAs so the async engine overlaps compute.
template <class F>
__device__ __forceinline__ void gemm_pipeline(int kchunks, u16* smem,
                                              v8f (&acc)[4][2], int lane,
                                              int waveM, int waveN, F&& fetch) {
  u16* bufA[2] = { smem,              smem + 2 * TILE_E };
  u16* bufB[2] = { smem + TILE_E,     smem + 3 * TILE_E };
  fetch(0, bufA[0], bufB[0]);
  async_wait_all();
  __syncthreads();
  for (int kc = 0; kc < kchunks; ++kc) {
    const int cur = kc & 1, nxt = cur ^ 1;
    if (kc + 1 < kchunks) fetch(kc + 1, bufA[nxt], bufB[nxt]);
    mma_tile(bufA[cur], bufB[cur], acc, lane, waveM, waveN);
    async_wait_all();
    __syncthreads();
  }
}

#define GEMM_PROLOG()                                                        \
  __shared__ u16 smem[4 * TILE_E];                                           \
  const int tid = threadIdx.x;                                               \
  const int lane = tid & 31, wave = tid >> 5;                                \
  const int waveM = wave & 1, waveN = wave >> 1;                             \
  const int m0 = blockIdx.y * 128, n0 = blockIdx.x * 128;                    \
  v8f acc[4][2] = {};

// ---------------------------------------------------------------------------
// init kernels
// ---------------------------------------------------------------------------
__global__ void k_convA(const float* __restrict__ A, u16* __restrict__ Abf) {
  const size_t i = (size_t)(blockIdx.x * 256 + threadIdx.x) * 4;
  float4 v = *(const float4*)(A + i);
  u32 lo = (u32)f2bf(v.x) | ((u32)f2bf(v.y) << 16);
  u32 hi = (u32)f2bf(v.z) | ((u32)f2bf(v.w) << 16);
  uint2 o; o.x = lo; o.y = hi;
  *(uint2*)(Abf + i) = o;
}

__global__ void k_initW(const float* Wgh, const float* Wgc,
                        const float* rk, const float* kk,
                        const float* Whc, const float* Whp,
                        const float* Wcc, const float* Wcp,
                        u16* WghT, u16* WgcT, u16* WzT, u16* WmhT, u16* WmcT) {
  int i = blockIdx.x * 256 + threadIdx.x;
  if (i < 65536) { int n = i >> 8, k = i & 255;        // W_graph_h^T [256][256]
    WghT[n * 256 + k] = f2bf(Wgh[k * 256 + n]); return; }
  i -= 65536;
  if (i < 65536) { int n = i >> 8, k = i & 255;        // W_graph_c^T
    WgcT[n * 256 + k] = f2bf(Wgc[k * 256 + n]); return; }
  i -= 65536;
  if (i < 294912) { int n = i / 288, k = i % 288;      // [rkernel;kernel;0]^T [1024][288]
    float v = (k < 256) ? rk[(size_t)k * 1024 + n]
            : (k < 256 + DD) ? kk[(size_t)(k - 256) * 1024 + n] : 0.0f;
    WzT[(size_t)n * 288 + k] = f2bf(v); return; }
  i -= 294912;
  if (i < 131072) { int n = i >> 9, k = i & 511;       // [W_h_cur;W_h_prev]^T [256][512]
    float v = (k < 256) ? Whc[k * 256 + n] : Whp[(k - 256) * 256 + n];
    WmhT[(size_t)n * 512 + k] = f2bf(v); return; }
  i -= 131072;
  if (i < 131072) { int n = i >> 9, k = i & 511;       // [W_c_cur;W_c_prev]^T
    float v = (k < 256) ? Wcc[k * 256 + n] : Wcp[(k - 256) * 256 + n];
    WmcT[(size_t)n * 512 + k] = f2bf(v); return; }
}

__global__ void k_xpad(const float* __restrict__ inputs, u16* __restrict__ xpad, int t) {
  const int i = blockIdx.x * 256 + threadIdx.x;        // 4096*32
  const int n = i >> 5, j = i & 31;
  xpad[i] = (j < DD) ? f2bf(inputs[((size_t)n * TT + t) * DD + j]) : (u16)0;
}

// ---------------------------------------------------------------------------
// XG = [h@Wgh+bgh | c@Wgc+bgc], stored TRANSPOSED (N-major) as the B-operand
// of the big graph GEMM.  M=4096 N=512 K=256.
// ---------------------------------------------------------------------------
__global__ __launch_bounds__(256)
void k_gates(const u16* __restrict__ h_bf, const u16* __restrict__ c_bf,
             const u16* __restrict__ WghT, const u16* __restrict__ WgcT,
             const float* __restrict__ bgh, const float* __restrict__ bgc,
             u16* __restrict__ XGT) {
  GEMM_PROLOG();
  const bool isC = (n0 >= 256);
  const u16* asrc = isC ? c_bf : h_bf;
  const u16* bsrc = isC ? WgcT : WghT;
  const int  bn0  = n0 & 255;
  gemm_pipeline(8, smem, acc, lane, waveM, waveN,
      [&](int kc, u16* dA, u16* dB) {
        fetch_tile(dA, asrc, HH, m0, kc * 32, tid);
        fetch_tile(dB, bsrc, HH, bn0, kc * 32, tid);
      });
  const float* bias = isC ? bgc : bgh;
  const int half = lane >> 4, ln = lane & 15;
#pragma unroll
  for (int mi = 0; mi < 4; ++mi)
#pragma unroll
    for (int ni = 0; ni < 2; ++ni) {
      const int n = n0 + waveN * 32 + ni * 16 + ln;
      const float bv = bias[n & 255];
#pragma unroll
      for (int i = 0; i < 8; ++i) {
        const int m = m0 + waveM * 64 + mi * 16 + i + 8 * half;
        XGT[(size_t)n * NN + m] = f2bf(acc[mi][ni][i] + bv);
      }
    }
}

// z = [h|x|0] @ Wz^T + bias   (M=4096 N=1024 K=288)
__global__ __launch_bounds__(256)
void k_z(const u16* __restrict__ h_bf, const u16* __restrict__ xpad,
         const u16* __restrict__ WzT, const float* __restrict__ lbias,
         float* __restrict__ z) {
  GEMM_PROLOG();
  gemm_pipeline(9, smem, acc, lane, waveM, waveN,
      [&](int kc, u16* dA, u16* dB) {
        if (kc < 8) fetch_tile(dA, h_bf, HH, m0, kc * 32, tid);
        else        fetch_tile(dA, xpad, 32, m0, 0, tid);
        fetch_tile(dB, WzT, 288, n0, kc * 32, tid);
      });
  const int half = lane >> 4, ln = lane & 15;
#pragma unroll
  for (int mi = 0; mi < 4; ++mi)
#pragma unroll
    for (int ni = 0; ni < 2; ++ni) {
      const int n = n0 + waveN * 32 + ni * 16 + ln;
      const float bv = lbias[n];
#pragma unroll
      for (int i = 0; i < 8; ++i) {
        const int m = m0 + waveM * 64 + mi * 16 + i + 8 * half;
        z[(size_t)m * 1024 + n] = acc[mi][ni][i] + bv;
      }
    }
}

// G = tanh(A_bf16 @ XG)   (M=4096 N=512 K=4096 — the heavy GEMM)
__global__ __launch_bounds__(256)
void k_graph(const u16* __restrict__ Abf, const u16* __restrict__ XGT,
             u16* __restrict__ G) {
  GEMM_PROLOG();
  gemm_pipeline(128, smem, acc, lane, waveM, waveN,
      [&](int kc, u16* dA, u16* dB) {
        if (kc + 1 < 128)   // keep streaming A ahead -> global_prefetch_b8
          __builtin_prefetch(Abf + (size_t)(m0 + (tid >> 1)) * NN + (kc + 1) * 32, 0, 1);
        fetch_tile(dA, Abf, NN, m0, kc * 32, tid);
        fetch_tile(dB, XGT, NN, n0, kc * 32, tid);
      });
  const int half = lane >> 4, ln = lane & 15;
#pragma unroll
  for (int mi = 0; mi < 4; ++mi)
#pragma unroll
    for (int ni = 0; ni < 2; ++ni) {
      const int n = n0 + waveN * 32 + ni * 16 + ln;
#pragma unroll
      for (int i = 0; i < 8; ++i) {
        const int m = m0 + waveM * 64 + mi * 16 + i + 8 * half;
        G[(size_t)m * 512 + n] = f2bf(tanhf(acc[mi][ni][i]));
      }
    }
}

// Keras LSTM cell elementwise (gate order i,f,g,o)
__global__ __launch_bounds__(256)
void k_lstm(const float* __restrict__ z, const float* __restrict__ c_f32,
            u16* __restrict__ h_cur, u16* __restrict__ c_cur) {
  const int idx = blockIdx.x * 256 + threadIdx.x;      // 4096*256
  const int n = idx >> 8, hh = idx & 255;
  const float* zr = z + (size_t)n * 1024;
  const float zi = zr[hh], zf = zr[256 + hh], zg = zr[512 + hh], zo = zr[768 + hh];
  const float cc = sigmoidf_(zf) * c_f32[idx] + sigmoidf_(zi) * tanhf(zg);
  const float hc = sigmoidf_(zo) * tanhf(cc);
  h_cur[idx] = f2bf(hc);
  c_cur[idx] = f2bf(cc);
}

// h_update / c_update: sigmoid([cur|graph] @ [W_cur;W_prev]^T + b)
// concat-K GEMM M=4096 N=256 K=512; blockIdx.z selects h (0) vs c (1).
__global__ __launch_bounds__(256)
void k_merge(const u16* __restrict__ h_cur, const u16* __restrict__ c_cur,
             const u16* __restrict__ G,
             const u16* __restrict__ WmhT, const u16* __restrict__ WmcT,
             const float* __restrict__ b_h, const float* __restrict__ b_c,
             u16* __restrict__ h_bf, u16* __restrict__ c_bf,
             float* __restrict__ c_f32) {
  GEMM_PROLOG();
  const int zsel = blockIdx.z;
  const u16* a1   = zsel ? c_cur : h_cur;
  const u16* g2   = G + zsel * 256;                    // h-graph / c-graph cols
  const u16* bsrc = zsel ? WmcT : WmhT;
  gemm_pipeline(16, smem, acc, lane, waveM, waveN,
      [&](int kc, u16* dA, u16* dB) {
        if (kc < 8) fetch_tile(dA, a1, HH, m0, kc * 32, tid);
        else        fetch_tile(dA, g2, 512, m0, (kc - 8) * 32, tid);
        fetch_tile(dB, bsrc, 512, n0, kc * 32, tid);
      });
  const float* bias = zsel ? b_c : b_h;
  const int half = lane >> 4, ln = lane & 15;
#pragma unroll
  for (int mi = 0; mi < 4; ++mi)
#pragma unroll
    for (int ni = 0; ni < 2; ++ni) {
      const int n = n0 + waveN * 32 + ni * 16 + ln;
      const float bv = bias[n];
#pragma unroll
      for (int i = 0; i < 8; ++i) {
        const int m = m0 + waveM * 64 + mi * 16 + i + 8 * half;
        const float v = sigmoidf_(acc[mi][ni][i] + bv);
        if (zsel == 0) {
          h_bf[(size_t)m * HH + n] = f2bf(v);
        } else {
          c_f32[(size_t)m * HH + n] = v;
          c_bf[(size_t)m * HH + n] = f2bf(v);
        }
      }
    }
}

// out[n,t,:] = h_update @ W_out + b_out   (P=2 -> per-row dot)
__global__ __launch_bounds__(256)
void k_out(const u16* __restrict__ h_bf, const float* __restrict__ W_out,
           const float* __restrict__ b_out, float* __restrict__ out, int t) {
  __shared__ float w[HH * 2];
  const int tid = threadIdx.x;
  for (int i = tid; i < HH * 2; i += 256) w[i] = W_out[i];
  __syncthreads();
  const int row = blockIdx.x * 256 + tid;
  const u16* hr = h_bf + (size_t)row * HH;
  float a0 = 0.f, a1 = 0.f;
#pragma unroll 4
  for (int h = 0; h < HH; ++h) {
    const float hv = bf2f(hr[h]);
    a0 += hv * w[2 * h];
    a1 += hv * w[2 * h + 1];
  }
  float* o = out + ((size_t)row * TT + t) * 2;
  o[0] = a0 + b_out[0];
  o[1] = a1 + b_out[1];
}

// ---------------------------------------------------------------------------
extern "C" void kernel_launch(void* const* d_in, const int* in_sizes, int n_in,
                              void* d_out, int out_size, void* d_ws, size_t ws_size,
                              hipStream_t stream) {
  (void)in_sizes; (void)n_in; (void)out_size; (void)ws_size;
  const float* inputs  = (const float*)d_in[0];
  const float* A       = (const float*)d_in[1];
  const float* kernelW = (const float*)d_in[2];
  const float* rkernel = (const float*)d_in[3];
  const float* lbias   = (const float*)d_in[4];
  const float* Wgh     = (const float*)d_in[5];
  const float* bgh     = (const float*)d_in[6];
  const float* Wgc     = (const float*)d_in[7];
  const float* bgc     = (const float*)d_in[8];
  const float* Whc     = (const float*)d_in[9];
  const float* Whp     = (const float*)d_in[10];
  const float* b_h     = (const float*)d_in[11];
  const float* Wcc     = (const float*)d_in[12];
  const float* Wcp     = (const float*)d_in[13];
  const float* b_c     = (const float*)d_in[14];
  const float* W_out   = (const float*)d_in[15];
  const float* b_out   = (const float*)d_in[16];
  float* out = (float*)d_out;

  char* p = (char*)d_ws;
  auto alloc = [&](size_t bytes) {
    char* r = p; p += (bytes + 255) & ~(size_t)255; return r;
  };
  u16*   Abf   = (u16*)  alloc((size_t)NN * NN * 2);     // 32 MB, L2-resident
  u16*   h_bf  = (u16*)  alloc((size_t)NN * HH * 2);
  u16*   c_bf  = (u16*)  alloc((size_t)NN * HH * 2);
  float* c_f   = (float*)alloc((size_t)NN * HH * 4);
  u16*   XGT   = (u16*)  alloc((size_t)512 * NN * 2);    // XG transposed
  u16*   G     = (u16*)  alloc((size_t)NN * 512 * 2);    // tanh graph states
  float* z     = (float*)alloc((size_t)NN * 1024 * 4);
  u16*   h_cur = (u16*)  alloc((size_t)NN * HH * 2);
  u16*   c_cur = (u16*)  alloc((size_t)NN * HH * 2);
  u16*   xpad  = (u16*)  alloc((size_t)NN * 32 * 2);
  u16*   WghT  = (u16*)  alloc((size_t)65536 * 2);
  u16*   WgcT  = (u16*)  alloc((size_t)65536 * 2);
  u16*   WzT   = (u16*)  alloc((size_t)294912 * 2);
  u16*   WmhT  = (u16*)  alloc((size_t)131072 * 2);
  u16*   WmcT  = (u16*)  alloc((size_t)131072 * 2);

  (void)hipMemsetAsync(h_bf, 0, (size_t)NN * HH * 2, stream);
  (void)hipMemsetAsync(c_bf, 0, (size_t)NN * HH * 2, stream);
  (void)hipMemsetAsync(c_f,  0, (size_t)NN * HH * 4, stream);

  k_convA<<<(NN * NN) / 4 / 256, 256, 0, stream>>>(A, Abf);
  const int initTotal = 65536 + 65536 + 294912 + 131072 + 131072;
  k_initW<<<(initTotal + 255) / 256, 256, 0, stream>>>(
      Wgh, Wgc, rkernel, kernelW, Whc, Whp, Wcc, Wcp,
      WghT, WgcT, WzT, WmhT, WmcT);

  for (int t = 0; t < TT; ++t) {
    k_xpad <<<(NN * 32) / 256, 256, 0, stream>>>(inputs, xpad, t);
    k_gates<<<dim3(4, 32),     256, 0, stream>>>(h_bf, c_bf, WghT, WgcT, bgh, bgc, XGT);
    k_z    <<<dim3(8, 32),     256, 0, stream>>>(h_bf, xpad, WzT, lbias, z);
    k_graph<<<dim3(4, 32),     256, 0, stream>>>(Abf, XGT, G);
    k_lstm <<<(NN * HH) / 256, 256, 0, stream>>>(z, c_f, h_cur, c_cur);
    k_merge<<<dim3(2, 32, 2),  256, 0, stream>>>(h_cur, c_cur, G, WmhT, WmcT,
                                                 b_h, b_c, h_bf, c_bf, c_f);
    k_out  <<<NN / 256,        256, 0, stream>>>(h_bf, W_out, b_out, out, t);
  }
}
